// MultiHeadAttention_46780783788231
// MI455X (gfx1250) — matrix operand order repaired and checked by reading, not compiled, not run on hardware
//
#include <hip/hip_runtime.h>
#include <hip/hip_bf16.h>

// ---------------------------------------------------------------------------
// MI455X (gfx1250) multi-head attention, bf16 WMMA everywhere.
//   B=8, S=1024, D=1024, H=16, DK=64, SCALE=0.04 (scores are divided by it).
// Pipeline: f32->bf16 cvt, 3x WMMA GEMM (QKV), flash-attention (WMMA QK^T and
// PV, online softmax), WMMA GEMM (+bias) for the output projection.
// Tile staging of the reused operands (weights / K tiles) goes through the
// Tensor Data Mover (tensor_load_to_lds, TENSORcnt) where available.
// ---------------------------------------------------------------------------

typedef __attribute__((ext_vector_type(16))) __bf16 v16bf;
typedef __attribute__((ext_vector_type(8)))  __bf16 v8bf;
typedef __attribute__((ext_vector_type(8)))  float  v8f;
typedef __attribute__((ext_vector_type(4)))  unsigned int u32x4;
typedef __attribute__((ext_vector_type(8)))  int i32x8;
typedef __attribute__((ext_vector_type(4)))  int i32x4;

#if defined(__AMDGCN__) && __has_builtin(__builtin_amdgcn_tensor_load_to_lds) && \
    __has_builtin(__builtin_amdgcn_s_wait_tensorcnt)
#define HAVE_TDM 1
#else
#define HAVE_TDM 0
#endif

static __device__ inline v8bf ld8(const __bf16* p) { return *(const v8bf*)p; }

static __device__ inline v16bf cat16(v8bf lo, v8bf hi) {
  v16bf r;
#pragma unroll
  for (int i = 0; i < 8; ++i) { r[i] = lo[i]; r[i + 8] = hi[i]; }
  return r;
}

static __device__ inline v8f wmma_bf16(v16bf a, v16bf b, v8f c) {
  // v_wmma_f32_16x16x32_bf16  D = A(16x32) * B(32x16) + C(16x16 f32)
  return __builtin_amdgcn_wmma_f32_16x16x32_bf16(
      /*neg_a=*/false, a, /*neg_b=*/false, b,
      /*c_mod=*/(short)0, c, /*reuse_a=*/false, /*reuse_b=*/false);
}

#if HAVE_TDM
// TDM 2D tile load: bf16 tensor [tensor_h x tensor_w] (row stride
// row_stride_elems), tile [tile_h x tile_w] starting at gaddr, into LDS at
// lds_addr with pad_amount DWORDs of padding every pad_interval-coded DWORDs
// (pad codes per CDNA5 ISA 8.4: interval 0=2dw..7=256dw, amount N = N+1 dw).
static __device__ inline void tdm_load_2d(unsigned int lds_addr,
                                          const void* gaddr, int tile_w,
                                          int tile_h, int tensor_w,
                                          int tensor_h, int row_stride_elems,
                                          int pad_interval_code,
                                          int pad_amount_code) {
  unsigned long long ga = (unsigned long long)(uintptr_t)gaddr;
  u32x4 g0;
  g0[0] = 1u;                                   // count=1, user descriptor
  g0[1] = lds_addr;                             // LDS byte address
  g0[2] = (unsigned int)ga;                     // global_addr[31:0]
  g0[3] = (unsigned int)((ga >> 32) & 0x1FFFFFFu) | (2u << 30);  // type=2
  i32x8 g1;
  g1[0] = (1 << 16)                             // data_size = 2 bytes
          | (1 << 20)                           // pad_enable
          | (pad_interval_code << 22) | (pad_amount_code << 25);
  g1[1] = (tensor_w & 0xFFFF) << 16;            // tensor_dim0[15:0]
  g1[2] = ((tensor_w >> 16) & 0xFFFF) | ((tensor_h & 0xFFFF) << 16);
  g1[3] = ((tensor_h >> 16) & 0xFFFF) | ((tile_w & 0xFFFF) << 16);
  g1[4] = tile_h & 0xFFFF;                      // tile_dim1 (tile_dim2 = 0)
  g1[5] = row_stride_elems;                     // tensor_dim0_stride[31:0]
  g1[6] = 0;
  g1[7] = 0;
  i32x4 z4 = {0, 0, 0, 0};
#if defined(__clang_major__) && __clang_major__ >= 23
  i32x8 z8 = {0, 0, 0, 0, 0, 0, 0, 0};
  __builtin_amdgcn_tensor_load_to_lds(g0, g1, z4, z4, z8, 0);
#else
  __builtin_amdgcn_tensor_load_to_lds(g0, g1, z4, z4, 0);
#endif
}
#endif  // HAVE_TDM

// ---------------------------------------------------------------------------
// Elementwise f32 -> bf16 (4 elems / thread, n multiple of 4 here)
// ---------------------------------------------------------------------------
__global__ __launch_bounds__(256) void f32_to_bf16_kernel(
    const float* __restrict__ in, __bf16* __restrict__ out, int n) {
  int i = (blockIdx.x * 256 + threadIdx.x) * 4;
  if (i + 3 < n) {
    float4 v = *(const float4*)&in[i];
    out[i + 0] = (__bf16)v.x;
    out[i + 1] = (__bf16)v.y;
    out[i + 2] = (__bf16)v.z;
    out[i + 3] = (__bf16)v.w;
  }
}

// ---------------------------------------------------------------------------
// WMMA GEMM: C[M,N] = A[M,K] * Bw[N,K]^T   (both bf16 row-major, f32 acc)
// Tile: 128(M) x 64(N), Ktile 32. 256 threads = 8 waves; wave w owns rows
// [w*16, w*16+16) and all 4 N-subtiles (4 accumulators). The B (weight)
// tile is staged by the TDM while all lanes stage the A tile.
// MODE 0: f32 out += bias (final projection, writes d_out)
// MODE 1: bf16 out, scattered to [B,H,S,DK] (QKV projections)
// ---------------------------------------------------------------------------
template <int MODE>
__global__ __launch_bounds__(256) void gemm_bf16_wmma(
    const __bf16* __restrict__ A, const __bf16* __restrict__ Bw,
    const float* __restrict__ bias, float* __restrict__ outF,
    __bf16* __restrict__ outQ, int M, int N, int K) {
  constexpr int TM = 128, TN = 64, TK = 32;
  constexpr int LDA = TK + 8;  // 40 elems = 80B row stride (16B aligned)
  __shared__ __bf16 sA[TM * LDA];
  __shared__ __bf16 sB[TN * LDA];

  const int tid  = threadIdx.x;
  const int wave = tid >> 5;
  const int lane = tid & 31;
  const int half = lane >> 4;   // wave32: which 16-lane half
  const int l16  = lane & 15;
  const int mBase = blockIdx.x * TM;
  const int nBase = blockIdx.y * TN;

  v8f acc[4] = {};

  for (int k0 = 0; k0 < K; k0 += TK) {
#if HAVE_TDM
    // ---- stage B tile 64x32 via Tensor Data Mover (pad 4 dw / 16 dw) ----
    if (wave == 0)
      tdm_load_2d((unsigned int)(uintptr_t)sB, &Bw[(size_t)nBase * K + k0],
                  /*tile_w=*/TK, /*tile_h=*/TN, /*tensor_w=*/K,
                  /*tensor_h=*/N, /*stride=*/K,
                  /*pad_interval(16dw)=*/3, /*pad_amount(4dw)=*/3);
#else
    {
      int row = tid >> 2;
      int col = (tid & 3) * 8;
      *(uint4*)&sB[row * LDA + col] =
          *(const uint4*)&Bw[(size_t)(nBase + row) * K + k0 + col];
    }
#endif
    // ---- stage A tile 128x32 (512 x 16B chunks, 2 per thread) ----
#pragma unroll
    for (int i = 0; i < 2; ++i) {
      int c = tid + i * 256;
      int row = c >> 2;
      int col = (c & 3) * 8;
      *(uint4*)&sA[row * LDA + col] =
          *(const uint4*)&A[(size_t)(mBase + row) * K + k0 + col];
    }
    if (k0 + TK < K) {  // global_prefetch_b8 of the next A tile
      __builtin_prefetch(&A[(size_t)(mBase + (tid >> 1)) * K + k0 + TK], 0, 1);
    }
#if HAVE_TDM
    if (wave == 0) __builtin_amdgcn_s_wait_tensorcnt(0);
#endif
    __syncthreads();

    // A fragment: lane l16 holds row M=l16; elems e -> K=half*16+e
    // (contiguous K, matching the B fragment)
    int arow = (wave * 16 + l16) * LDA;
    v16bf afrag = cat16(ld8(&sA[arow + half * 16]),
                        ld8(&sA[arow + half * 16 + 8]));
#pragma unroll
    for (int nt = 0; nt < 4; ++nt) {
      // B fragment: lane = N + 16*(K>=16); elems are 16 contiguous K.
      int brow = (nt * 16 + l16) * LDA;
      v16bf bfrag = cat16(ld8(&sB[brow + half * 16]),
                          ld8(&sB[brow + half * 16 + 8]));
      acc[nt] = wmma_bf16(afrag, bfrag, acc[nt]);
    }
    __syncthreads();
  }

  // epilogue: C layout — VGPR r: lanes 0-15 (M=r,N=lane), 16-31 (M=r+8)
#pragma unroll
  for (int nt = 0; nt < 4; ++nt) {
#pragma unroll
    for (int r = 0; r < 8; ++r) {
      int row = mBase + wave * 16 + r + half * 8;
      int col = nBase + nt * 16 + l16;
      float v = acc[nt][r];
      if (MODE == 0) {
        outF[(size_t)row * N + col] = v + bias[col];
      } else {
        int b = row >> 10, s = row & 1023;   // M = B*S
        int h = col >> 6, dk = col & 63;     // N = H*DK
        outQ[((((size_t)b * 16 + h) * 1024 + s) << 6) + dk] = (__bf16)v;
      }
    }
  }
}

// ---------------------------------------------------------------------------
// Flash attention per (b,h): grid = (S/64, B*H), 128 threads (4 waves).
// Wave w owns 16 query rows. K tiles of 64 keys staged by the TDM; V tiles
// transposed manually into LDS. Online softmax with the reference's
// where(s==0,-1e9) mask and 1/SCALE = 25 scaling.
// ---------------------------------------------------------------------------
__global__ __launch_bounds__(128) void flash_attn_wmma(
    const __bf16* __restrict__ Q, const __bf16* __restrict__ Kk,
    const __bf16* __restrict__ V, __bf16* __restrict__ Z) {
  constexpr float INV_SCALE = 25.0f;  // 1/0.04
  constexpr int LD = 72;              // 144B row stride, 16B aligned
  __shared__ __bf16 skt[64 * LD];     // K tile   [t][dk]
  __shared__ __bf16 svt[64 * LD];     // V tile transposed [dk][t]
  __shared__ __bf16 sp[64 * LD];      // P tile   [q][t]

  const int tid = threadIdx.x;
  const int wave = tid >> 5, lane = tid & 31;
  const int half = lane >> 4, l16 = lane & 15;
  const int bh = blockIdx.y;                     // b*16 + h
  const int b = bh >> 4, h = bh & 15;
  const int qBase = blockIdx.x * 64;
  const size_t headOff = (size_t)bh * 1024 * 64;

  // Q A-fragments live in registers for the whole kernel (DK=64 -> 2 frags)
  const __bf16* qptr = Q + headOff + (size_t)(qBase + wave * 16 + l16) * 64;
  v16bf aq0 = cat16(*(const v8bf*)&qptr[half * 16],
                    *(const v8bf*)&qptr[half * 16 + 8]);
  v16bf aq1 = cat16(*(const v8bf*)&qptr[32 + half * 16],
                    *(const v8bf*)&qptr[32 + half * 16 + 8]);

  float m_i[8], l_i[8];
  v8f o[4] = {};
#pragma unroll
  for (int r = 0; r < 8; ++r) { m_i[r] = -1e30f; l_i[r] = 0.0f; }

  for (int t0 = 0; t0 < 1024; t0 += 64) {
    __syncthreads();
#if HAVE_TDM
    // ---- K tile 64x64 via Tensor Data Mover (pad 4 dw / 32 dw) ----
    if (wave == 0)
      tdm_load_2d((unsigned int)(uintptr_t)skt,
                  Kk + headOff + (size_t)t0 * 64,
                  /*tile_w=*/64, /*tile_h=*/64, /*tensor_w=*/64,
                  /*tensor_h=*/1024, /*stride=*/64,
                  /*pad_interval(32dw)=*/4, /*pad_amount(4dw)=*/3);
#else
    {
      int row = tid >> 1, co = (tid & 1) * 32;
      const __bf16* ksrc = Kk + headOff + (size_t)(t0 + row) * 64 + co;
      *(uint4*)&skt[row * LD + co + 0]  = *(const uint4*)&ksrc[0];
      *(uint4*)&skt[row * LD + co + 8]  = *(const uint4*)&ksrc[8];
      *(uint4*)&skt[row * LD + co + 16] = *(const uint4*)&ksrc[16];
      *(uint4*)&skt[row * LD + co + 24] = *(const uint4*)&ksrc[24];
    }
#endif
    // ---- stage V tile transposed: svt[dk][t] ----
    {
      int row = tid >> 1, co = (tid & 1) * 32;
      const __bf16* vsrc = V + headOff + (size_t)(t0 + row) * 64 + co;
      v8bf v0 = ld8(&vsrc[0]), v1 = ld8(&vsrc[8]);
      v8bf v2 = ld8(&vsrc[16]), v3 = ld8(&vsrc[24]);
#pragma unroll
      for (int i = 0; i < 8; ++i) {
        svt[(co + i + 0)  * LD + row] = v0[i];
        svt[(co + i + 8)  * LD + row] = v1[i];
        svt[(co + i + 16) * LD + row] = v2[i];
        svt[(co + i + 24) * LD + row] = v3[i];
      }
    }
#if HAVE_TDM
    if (wave == 0) __builtin_amdgcn_s_wait_tensorcnt(0);
#endif
    __syncthreads();

    // ---- scores S = Q K^T for 4 key sub-tiles of 16 ----
    v8f sacc[4];
#pragma unroll
    for (int nt = 0; nt < 4; ++nt) {
      int trow = (nt * 16 + l16) * LD;
      v16bf bk0 = cat16(ld8(&skt[trow + half * 16]),
                        ld8(&skt[trow + half * 16 + 8]));
      v16bf bk1 = cat16(ld8(&skt[trow + 32 + half * 16]),
                        ld8(&skt[trow + 32 + half * 16 + 8]));
      v8f z8 = {};
      z8 = wmma_bf16(aq0, bk0, z8);
      z8 = wmma_bf16(aq1, bk1, z8);
      sacc[nt] = z8;
    }

    // ---- mask + scale (reference: where(s==0,-1e9) then /SCALE) ----
#pragma unroll
    for (int nt = 0; nt < 4; ++nt)
#pragma unroll
      for (int r = 0; r < 8; ++r) {
        float s = sacc[nt][r];
        s = (s == 0.0f) ? -1e9f : s;
        sacc[nt][r] = s * INV_SCALE;
      }

    // ---- online softmax per row (rows striped: VGPR r <-> row r+8*half;
    //      the 16 lanes of a half hold the 16 columns -> xor-reduce <16) ----
#pragma unroll
    for (int r = 0; r < 8; ++r) {
      float mx = fmaxf(fmaxf(sacc[0][r], sacc[1][r]),
                       fmaxf(sacc[2][r], sacc[3][r]));
#pragma unroll
      for (int msk = 8; msk >= 1; msk >>= 1)
        mx = fmaxf(mx, __shfl_xor(mx, msk, 32));
      float mnew = fmaxf(m_i[r], mx);
      float alpha = __expf(m_i[r] - mnew);
#pragma unroll
      for (int nt = 0; nt < 4; ++nt) o[nt][r] *= alpha;
      float rsum = 0.0f;
#pragma unroll
      for (int nt = 0; nt < 4; ++nt) {
        float p = __expf(sacc[nt][r] - mnew);
        sacc[nt][r] = p;
        rsum += p;
      }
#pragma unroll
      for (int msk = 8; msk >= 1; msk >>= 1)
        rsum += __shfl_xor(rsum, msk, 32);
      l_i[r] = l_i[r] * alpha + rsum;
      m_i[r] = mnew;
    }

    // ---- restage P (C layout -> A layout) through LDS; wave-private rows,
    //      LDS is in-order per wave so s_wait_dscnt suffices (no barrier) ----
#pragma unroll
    for (int nt = 0; nt < 4; ++nt)
#pragma unroll
      for (int r = 0; r < 8; ++r)
        sp[(wave * 16 + r + half * 8) * LD + nt * 16 + l16] =
            (__bf16)sacc[nt][r];
    asm volatile("s_wait_dscnt 0x0" ::: "memory");

    int prow = (wave * 16 + l16) * LD;
    v16bf ap0 = cat16(ld8(&sp[prow + half * 16]),
                      ld8(&sp[prow + half * 16 + 8]));
    v16bf ap1 = cat16(ld8(&sp[prow + 32 + half * 16]),
                      ld8(&sp[prow + 32 + half * 16 + 8]));

    // ---- O += P * V  (B frags from transposed V: contiguous keys) ----
#pragma unroll
    for (int nt = 0; nt < 4; ++nt) {
      int vrow = (nt * 16 + l16) * LD;
      v16bf bv0 = cat16(ld8(&svt[vrow + half * 16]),
                        ld8(&svt[vrow + half * 16 + 8]));
      v16bf bv1 = cat16(ld8(&svt[vrow + 32 + half * 16]),
                        ld8(&svt[vrow + 32 + half * 16 + 8]));
      o[nt] = wmma_bf16(ap0, bv0, o[nt]);
      o[nt] = wmma_bf16(ap1, bv1, o[nt]);
    }
  }

  // ---- normalize and write z as bf16 [B, S, H*DK] for the out projection ----
#pragma unroll
  for (int nt = 0; nt < 4; ++nt)
#pragma unroll
    for (int r = 0; r < 8; ++r) {
      int s = qBase + wave * 16 + r + half * 8;
      int dk = nt * 16 + l16;
      float v = o[nt][r] / l_i[r];
      Z[((size_t)b * 1024 + s) * 1024 + h * 64 + dk] = (__bf16)v;
    }
}

// ---------------------------------------------------------------------------
// Launcher
// ---------------------------------------------------------------------------
extern "C" void kernel_launch(void* const* d_in, const int* in_sizes, int n_in,
                              void* d_out, int out_size, void* d_ws,
                              size_t ws_size, hipStream_t stream) {
  constexpr int B = 8, S = 1024, D = 1024, H = 16, DK = 64;
  constexpr int M = B * S;        // 8192
  constexpr int N = H * DK;       // 1024

  const float* x  = (const float*)d_in[0];
  const float* Wq = (const float*)d_in[1];
  const float* Wk = (const float*)d_in[2];
  const float* Wv = (const float*)d_in[3];
  const float* Wo = (const float*)d_in[4];
  const float* bo = (const float*)d_in[5];

  char* w = (char*)d_ws;
  __bf16* xb  = (__bf16*)w; w += (size_t)M * D * 2;        // 16 MB
  __bf16* wqb = (__bf16*)w; w += (size_t)N * D * 2;        //  2 MB
  __bf16* wkb = (__bf16*)w; w += (size_t)N * D * 2;
  __bf16* wvb = (__bf16*)w; w += (size_t)N * D * 2;
  __bf16* wob = (__bf16*)w; w += (size_t)D * N * 2;
  __bf16* qb  = (__bf16*)w; w += (size_t)B * H * S * DK * 2;  // 16 MB each
  __bf16* kb  = (__bf16*)w; w += (size_t)B * H * S * DK * 2;
  __bf16* vb  = (__bf16*)w; w += (size_t)B * H * S * DK * 2;
  __bf16* zb  = (__bf16*)w; w += (size_t)M * N * 2;

  // 1) f32 -> bf16 conversions
  f32_to_bf16_kernel<<<(M * D / 4 + 255) / 256, 256, 0, stream>>>(x, xb, M * D);
  f32_to_bf16_kernel<<<(N * D / 4 + 255) / 256, 256, 0, stream>>>(Wq, wqb, N * D);
  f32_to_bf16_kernel<<<(N * D / 4 + 255) / 256, 256, 0, stream>>>(Wk, wkb, N * D);
  f32_to_bf16_kernel<<<(N * D / 4 + 255) / 256, 256, 0, stream>>>(Wv, wvb, N * D);
  f32_to_bf16_kernel<<<(D * N / 4 + 255) / 256, 256, 0, stream>>>(Wo, wob, D * N);

  // 2) QKV projections (scatter to [B,H,S,DK] bf16)
  dim3 gGrid(M / 128, N / 64);
  gemm_bf16_wmma<1><<<gGrid, 256, 0, stream>>>(xb, wqb, nullptr, nullptr, qb, M, N, D);
  gemm_bf16_wmma<1><<<gGrid, 256, 0, stream>>>(xb, wkb, nullptr, nullptr, kb, M, N, D);
  gemm_bf16_wmma<1><<<gGrid, 256, 0, stream>>>(xb, wvb, nullptr, nullptr, vb, M, N, D);

  // 3) flash attention: (S/64, B*H) workgroups of 4 waves
  flash_attn_wmma<<<dim3(S / 64, B * H), 128, 0, stream>>>(qb, kb, vb, zb);

  // 4) output projection: d_out = z @ Wo^T + bo (f32)
  gemm_bf16_wmma<0><<<gGrid, 256, 0, stream>>>(zb, wob, bo, (float*)d_out,
                                               nullptr, M, D, N);
}
